// MultiHeadSelfAttention_62405874811098
// MI455X (gfx1250) — matrix operand order repaired and checked by reading, not compiled, not run on hardware
//
#include <hip/hip_runtime.h>
#include <hip/hip_bf16.h>

typedef __attribute__((ext_vector_type(16))) _Float16 v16h;
typedef __attribute__((ext_vector_type(8)))  _Float16 v8h;
typedef __attribute__((ext_vector_type(8)))  float    v8f;
typedef __attribute__((ext_vector_type(4)))  int      v4i;

#define WMMA_F32_F16(a, b, c) \
  __builtin_amdgcn_wmma_f32_16x16x32_f16(false, (a), false, (b), (short)0, (c), false, false)

#if defined(__has_builtin)
#  if __has_builtin(__builtin_amdgcn_global_load_async_to_lds_b128)
#    define HAVE_ASYNC_LDS 1
#  endif
#  if __has_builtin(__builtin_amdgcn_s_wait_asynccnt)
#    define HAVE_WAIT_ASYNC 1
#  endif
#endif

// ---- async global->LDS staging (CDNA5 path; fallback = through-VGPR copy) --
__device__ inline void async_ld16(const _Float16* g, _Float16* l) {
#if defined(HAVE_ASYNC_LDS)
  typedef __attribute__((address_space(1))) v4i* gptr_t;
  typedef __attribute__((address_space(3))) v4i* lptr_t;
  __builtin_amdgcn_global_load_async_to_lds_b128(
      (gptr_t)(v4i*)(g), (lptr_t)(v4i*)(l), 0, 0);
#else
  *(uint4*)l = *(const uint4*)g;
#endif
}
template <int N>
__device__ inline void wait_async() {
#if defined(HAVE_WAIT_ASYNC)
  __builtin_amdgcn_s_wait_asynccnt(N);
#elif defined(HAVE_ASYNC_LDS)
  asm volatile("s_wait_asynccnt %0" :: "n"(N) : "memory");
#endif
}

// ---- fragment helpers (contiguous k order: k = 16*(lane/16) + s) -------------------------------
__device__ inline v16h make_afrag(const _Float16* p_lo, const _Float16* p_hi) {
  union { v16h v; v8h h[2]; } u;
  u.h[0] = *(const v8h*)p_lo;
  u.h[1] = *(const v8h*)p_hi;
  return u.v;
}
__device__ inline v16h make_bfrag(const _Float16* p) {
  union { v16h v; v8h h[2]; } u;
  u.h[0] = *(const v8h*)p;
  u.h[1] = *(const v8h*)(p + 8);
  return u.v;
}

// ---- kernel 1: LayerNorm -> f16 ------------------------------------------
__global__ __launch_bounds__(256) void ln_to_f16(const float* __restrict__ X,
                                                 const float* __restrict__ gamma,
                                                 const float* __restrict__ beta,
                                                 _Float16* __restrict__ Xn) {
  const int row = blockIdx.x;
  const float* x = X + (size_t)row * 1024;
  __shared__ float red[2][8];
  float s = 0.f, ss = 0.f;
  for (int i = threadIdx.x; i < 1024; i += 256) { float v = x[i]; s += v; ss += v * v; }
  #pragma unroll
  for (int off = 1; off < 32; off <<= 1) { s += __shfl_xor(s, off, 32); ss += __shfl_xor(ss, off, 32); }
  const int w = threadIdx.x >> 5, lane = threadIdx.x & 31;
  if (lane == 0) { red[0][w] = s; red[1][w] = ss; }
  __syncthreads();
  float S1 = 0.f, S2 = 0.f;
  #pragma unroll
  for (int i = 0; i < 8; ++i) { S1 += red[0][i]; S2 += red[1][i]; }
  const float mu   = S1 * (1.0f / 1024.0f);
  const float var  = S2 * (1.0f / 1024.0f) - mu * mu;
  const float rstd = rsqrtf(var + 1e-5f);
  _Float16* o = Xn + (size_t)row * 1024;
  for (int i = threadIdx.x; i < 1024; i += 256)
    o[i] = (_Float16)((x[i] - mu) * rstd * gamma[i] + beta[i]);
}

// ---- kernel 2: weights f32 -> f16 ----------------------------------------
__global__ __launch_bounds__(256) void cvt_weights(const float* __restrict__ a,
                                                   const float* __restrict__ b,
                                                   const float* __restrict__ c,
                                                   const float* __restrict__ d,
                                                   _Float16* __restrict__ o) {
  const int idx = blockIdx.x * 256 + threadIdx.x;
  const int which = idx >> 20;
  const int off   = idx & 0xFFFFF;
  const float* src = (which == 0) ? a : (which == 1) ? b : (which == 2) ? c : d;
  o[idx] = (_Float16)src[off];
}

// ---- kernel 3/5: WMMA GEMM  C = A[M,1024] * W[N,1024]^T + bias -----------
// Double-buffered async-to-LDS pipeline, ONE barrier per k-step:
//   [stage k+1 into buf^1] -> [wmma from buf] -> [wait_async + barrier]
// MODE 0: f16 out, head-major [4,16,2048,64], scaled.  MODE 1: f32 row-major.
template <int MODE>
__global__ __launch_bounds__(256) void gemm_wmma(const _Float16* __restrict__ A,
                                                 const _Float16* __restrict__ W,
                                                 const float* __restrict__ bias,
                                                 void* __restrict__ Cout,
                                                 float scale) {
  constexpr int K = 1024;
  constexpr int NT = K / 32;
  __shared__ alignas(16) _Float16 As[2][128][40];
  __shared__ alignas(16) _Float16 Bs[2][128][40];
  const int m0 = blockIdx.y * 128, n0 = blockIdx.x * 128;
  const int tid = threadIdx.x, lane = tid & 31, wid = tid >> 5;
  const int wm = wid >> 1, wn = wid & 1;
  const int lg = lane >> 4, lm = lane & 15;

  v8f acc[2][4] = {};

  // stage one 128x32 A-tile + 128x32 B-tile: 4 async b128 ops per thread
  auto stage = [&](int buf, int k0) {
    #pragma unroll
    for (int i = 0; i < 2; ++i) {
      const int e = i * 2048 + tid * 8;          // half index within 128x32 tile
      const int r = e >> 5, c = e & 31;
      async_ld16(&A[(size_t)(m0 + r) * K + k0 + c], &As[buf][r][c]);
      async_ld16(&W[(size_t)(n0 + r) * K + k0 + c], &Bs[buf][r][c]);
    }
  };

  stage(0, 0);
  wait_async<0>();
  __syncthreads();

  for (int kt = 0; kt < NT; ++kt) {
    const int cur = kt & 1;
    const bool more = (kt + 1) < NT;
    if (more) stage(cur ^ 1, (kt + 1) * 32);   // overlaps with this step's WMMAs

    v16h af[2], bf[4];
    #pragma unroll
    for (int sm = 0; sm < 2; ++sm) {
      const int m = wm * 32 + sm * 16 + lm;
      af[sm] = make_afrag(&As[cur][m][lg * 16], &As[cur][m][lg * 16 + 8]);
    }
    #pragma unroll
    for (int sn = 0; sn < 4; ++sn) {
      const int n = wn * 64 + sn * 16 + lm;
      bf[sn] = make_bfrag(&Bs[cur][n][lg * 16]);
    }
    #pragma unroll
    for (int sm = 0; sm < 2; ++sm)
      #pragma unroll
      for (int sn = 0; sn < 4; ++sn)
        acc[sm][sn] = WMMA_F32_F16(af[sm], bf[sn], acc[sm][sn]);

    if (more) {
      wait_async<0>();            // next buffer staged (my 4 ops)
      __syncthreads();            // all waves staged + done reading buf[cur]
    }
  }

  #pragma unroll
  for (int sm = 0; sm < 2; ++sm)
    #pragma unroll
    for (int sn = 0; sn < 4; ++sn)
      #pragma unroll
      for (int r = 0; r < 8; ++r) {
        const int gm = m0 + wm * 32 + sm * 16 + r + 8 * lg;
        const int gn = n0 + wn * 64 + sn * 16 + lm;
        const float v = (acc[sm][sn][r] + bias[gn]) * scale;
        if (MODE == 0) {
          _Float16* O = (_Float16*)Cout;
          const int b = gm >> 11, s = gm & 2047, h = gn >> 6, d = gn & 63;
          O[((size_t)((b * 16 + h) * 2048 + s)) * 64 + d] = (_Float16)v;
        } else {
          float* O = (float*)Cout;
          O[(size_t)gm * 1024 + gn] = v;
        }
      }
}

// ---- kernel 4: causal flash attention ------------------------------------
// grid: (32 q-tiles, B*H=64). block: 128 threads = 4 waves, 16 q-rows/wave.
// K/V double-buffered: stage block kb+1 while computing block kb; one
// barrier per iteration.
__global__ __launch_bounds__(128) void flash_attn(const _Float16* __restrict__ Q,
                                                  const _Float16* __restrict__ Kt,
                                                  const _Float16* __restrict__ V,
                                                  _Float16* __restrict__ Mrg) {
  constexpr int S = 2048, Dh = 64;
  const int qt = blockIdx.x;
  const int bh = blockIdx.y;
  const size_t base = (size_t)bh * S * Dh;

  __shared__ alignas(16) _Float16 Qs[64][72];
  __shared__ alignas(16) _Float16 Ks[2][32][72];
  __shared__ alignas(16) _Float16 Vts[2][64][40];   // Vts[.][d][k]
  __shared__ alignas(16) _Float16 Ps[4][16][40];

  const int tid = threadIdx.x, lane = tid & 31, w = tid >> 5;
  const int lg = lane >> 4, lm = lane & 15;

  // stage one K (async) + V (manual transpose) 32-key block
  auto stage_kv = [&](int buf, int k0) {
    #pragma unroll
    for (int i = 0; i < 2; ++i) {
      const int e = i * 1024 + tid * 8;
      const int r = e >> 6, c = e & 63;
      async_ld16(Kt + base + (size_t)(k0 + r) * Dh + c, &Ks[buf][r][c]);
    }
    const int kk = tid & 31, dg = tid >> 5;
    const _Float16* src = V + base + (size_t)(k0 + kk) * Dh + dg * 16;
    #pragma unroll
    for (int j = 0; j < 16; ++j) Vts[buf][dg * 16 + j][kk] = src[j];
  };

  // stage Q tile 64x64 via async-to-LDS
  #pragma unroll
  for (int i = 0; i < 4; ++i) {
    const int e = i * 1024 + tid * 8;
    const int r = e >> 6, c = e & 63;
    async_ld16(Q + base + (size_t)(qt * 64 + r) * Dh + c, &Qs[r][c]);
  }
  stage_kv(0, 0);
  wait_async<0>();
  __syncthreads();

  // Q fragments are loop-invariant: hoist (2 frags covering Dh=64)
  v16h aq[2];
  #pragma unroll
  for (int d = 0; d < 2; ++d)
    aq[d] = make_afrag(&Qs[w * 16 + lm][d * 32 + lg * 16],
                       &Qs[w * 16 + lm][d * 32 + lg * 16 + 8]);

  float mrow[8], lrow[8];
  v8f   oacc[4] = {};
  #pragma unroll
  for (int r = 0; r < 8; ++r) { mrow[r] = -3.0e38f; lrow[r] = 0.0f; }

  const int wave_row0 = qt * 64 + w * 16;
  const int nkb = 2 * qt + 2;

  for (int kb = 0; kb < nkb; ++kb) {
    const int k0 = kb * 32;
    const int cur = kb & 1;
    const bool more = (kb + 1) < nkb;
    if (more) stage_kv(cur ^ 1, k0 + 32);     // overlaps with this block's math

    if (k0 <= wave_row0 + 15) {
      v8f sacc[2] = {};
      #pragma unroll
      for (int d = 0; d < 2; ++d)
        #pragma unroll
        for (int kk2 = 0; kk2 < 2; ++kk2) {
          const v16h bk = make_bfrag(&Ks[cur][kk2 * 16 + lm][d * 32 + lg * 16]);
          sacc[kk2] = WMMA_F32_F16(aq[d], bk, sacc[kk2]);
        }

      // causal mask only needed for the diagonal block of this wave
      if ((k0 + 31) > wave_row0) {
        #pragma unroll
        for (int r = 0; r < 8; ++r) {
          const int grow = wave_row0 + r + 8 * lg;
          #pragma unroll
          for (int kk2 = 0; kk2 < 2; ++kk2)
            if (k0 + kk2 * 16 + lm > grow) sacc[kk2][r] = -3.0e38f;
        }
      }

      float alpha[8];
      #pragma unroll
      for (int r = 0; r < 8; ++r) {
        float mb = fmaxf(sacc[0][r], sacc[1][r]);
        #pragma unroll
        for (int off = 1; off < 16; off <<= 1) mb = fmaxf(mb, __shfl_xor(mb, off, 32));
        const float mn = fmaxf(mrow[r], mb);
        alpha[r] = __expf(mrow[r] - mn);
        mrow[r]  = mn;
        float rs = 0.0f;
        #pragma unroll
        for (int kk2 = 0; kk2 < 2; ++kk2) {
          const float p = __expf(sacc[kk2][r] - mn);
          rs += p;
          Ps[w][r + 8 * lg][kk2 * 16 + lm] = (_Float16)p;
        }
        #pragma unroll
        for (int off = 1; off < 16; off <<= 1) rs += __shfl_xor(rs, off, 32);
        lrow[r] = lrow[r] * alpha[r] + rs;
      }

      #pragma unroll
      for (int sn = 0; sn < 4; ++sn)
        #pragma unroll
        for (int r = 0; r < 8; ++r) oacc[sn][r] *= alpha[r];

      const v16h ap = make_afrag(&Ps[w][lm][lg * 16], &Ps[w][lm][lg * 16 + 8]);
      #pragma unroll
      for (int sn = 0; sn < 4; ++sn) {
        const v16h bv = make_bfrag(&Vts[cur][sn * 16 + lm][lg * 16]);
        oacc[sn] = WMMA_F32_F16(ap, bv, oacc[sn]);
      }
    }

    if (more) {
      wait_async<0>();          // my async K loads for buf[cur^1] complete
      __syncthreads();          // all waves staged + done reading buf[cur]
    }
  }

  const int b = bh >> 4, h = bh & 15;
  #pragma unroll
  for (int sn = 0; sn < 4; ++sn) {
    #pragma unroll
    for (int r = 0; r < 8; ++r) {
      const float inv = 1.0f / lrow[r];
      const int grow = wave_row0 + r + 8 * lg;
      const int d    = sn * 16 + lm;
      Mrg[((size_t)(b * 2048 + grow)) * 1024 + h * 64 + d] = (_Float16)(oacc[sn][r] * inv);
    }
  }
}

// ---- host launcher -------------------------------------------------------
extern "C" void kernel_launch(void* const* d_in, const int* in_sizes, int n_in,
                              void* d_out, int out_size, void* d_ws, size_t ws_size,
                              hipStream_t stream) {
  const float* X        = (const float*)d_in[0];
  const float* ln_gamma = (const float*)d_in[1];
  const float* ln_beta  = (const float*)d_in[2];
  const float* Wq = (const float*)d_in[3];  const float* bq = (const float*)d_in[4];
  const float* Wk = (const float*)d_in[5];  const float* bk = (const float*)d_in[6];
  const float* Wv = (const float*)d_in[7];  const float* bv = (const float*)d_in[8];
  const float* Wo = (const float*)d_in[9];  const float* bo = (const float*)d_in[10];

  char* ws = (char*)d_ws;
  const size_t MB = 1024ull * 1024ull;
  _Float16* Xn16 = (_Float16*)(ws);                 // 16 MB
  _Float16* W16  = (_Float16*)(ws + 16 * MB);       //  8 MB
  _Float16* Q16  = (_Float16*)(ws + 24 * MB);       // 16 MB
  _Float16* K16  = (_Float16*)(ws + 40 * MB);       // 16 MB
  _Float16* V16  = (_Float16*)(ws + 56 * MB);       // 16 MB
  _Float16* M16  = (_Float16*)(ws + 72 * MB);       // 16 MB
  _Float16* Wq16 = W16;
  _Float16* Wk16 = W16 + 1 * MB;
  _Float16* Wv16 = W16 + 2 * MB;
  _Float16* Wo16 = W16 + 3 * MB;

  ln_to_f16<<<8192, 256, 0, stream>>>(X, ln_gamma, ln_beta, Xn16);
  cvt_weights<<<16384, 256, 0, stream>>>(Wq, Wk, Wv, Wo, W16);
  dim3 ggrid(8, 64);
  gemm_wmma<0><<<ggrid, 256, 0, stream>>>(Xn16, Wq16, bq, (void*)Q16, 0.125f);
  gemm_wmma<0><<<ggrid, 256, 0, stream>>>(Xn16, Wk16, bk, (void*)K16, 1.0f);
  gemm_wmma<0><<<ggrid, 256, 0, stream>>>(Xn16, Wv16, bv, (void*)V16, 1.0f);
  flash_attn<<<dim3(32, 64), 128, 0, stream>>>(Q16, K16, V16, M16);
  gemm_wmma<1><<<ggrid, 256, 0, stream>>>(M16, Wo16, bo, d_out, 1.0f);
}
